// OnlineFlashMTPModel_5609227289075
// MI455X (gfx1250) — matrix operand order repaired and checked by reading, not compiled, not run on hardware
//
#include <hip/hip_runtime.h>
#include <hip/hip_bf16.h>
#include <stdint.h>

// Problem constants (from reference)
#define L_   4
#define B_   2
#define S_   4096
#define H_   2048
#define N_   256
#define BS_  16
#define NH_  16
#define DH_  128
#define SCALE_ 0.08838834764831845f   // 1/sqrt(128)

// GEMM dims: C[GM][GN] = A[GM][GK] * W[GK][GN]
#define GM 2048   // B*N*L gathered rows
#define GK 2048   // H
#define GN 4096   // [W_kc | W_vc] columns (k_chs | v_chs)

typedef __attribute__((ext_vector_type(16))) __bf16 v16bf;
typedef __attribute__((ext_vector_type(8)))  __bf16 v8bf;
typedef __attribute__((ext_vector_type(8)))  float  v8f;

__device__ __forceinline__ unsigned short f32_to_bf16_rne(float f) {
    union { float f; unsigned int u; } cv; cv.f = f;
    unsigned int u = cv.u;
    unsigned int r = u + 0x7fffu + ((u >> 16) & 1u);   // round-to-nearest-even
    return (unsigned short)(r >> 16);
}

// ---------------------------------------------------------------------------
// K1: transpose + convert [W_kc | W_vc] (f32, row-major [k][c]) into
//     Wt bf16 column-major [c][k]  (c in [0,4096), k in [0,2048))
// ---------------------------------------------------------------------------
__global__ __launch_bounds__(256)
void wt_transpose_kernel(const float* __restrict__ Wkc,
                         const float* __restrict__ Wvc,
                         unsigned short* __restrict__ Wt) {
    __shared__ float tile[32][33];            // +1 pad: no bank conflicts
    const int ct = blockIdx.x;                // 0..127 : 32-col tile over 4096
    const int kt = blockIdx.y;                // 0..63  : 32-row tile over 2048
    const int tx = threadIdx.x;               // 0..31
    const int ty = threadIdx.y;               // 0..7
    const int cbase = ct * 32, kbase = kt * 32;
    const float* W = (cbase < 2048) ? Wkc : Wvc;
    const int coff = (cbase < 2048) ? cbase : (cbase - 2048);

#pragma unroll
    for (int i = 0; i < 4; ++i) {             // coalesced f32 reads along c
        int k = kbase + ty + i * 8;
        tile[ty + i * 8][tx] = W[(size_t)k * 2048 + coff + tx];
    }
    __syncthreads();
#pragma unroll
    for (int i = 0; i < 4; ++i) {             // contiguous bf16 writes along k
        int c = cbase + ty + i * 8;
        Wt[(size_t)c * GK + kbase + tx] = f32_to_bf16_rne(tile[tx][ty + i * 8]);
    }
}

// ---------------------------------------------------------------------------
// K2: gather chs rows and convert to bf16 A[GM][GK]
//     row m = ((b*N + n)*L + l)  ->  hidden_states[l, b, clip(anchor-1,0), :]
// ---------------------------------------------------------------------------
__global__ __launch_bounds__(256)
void gather_a_kernel(const float* __restrict__ hs,
                     const long long* __restrict__ anchor,
                     unsigned short* __restrict__ A) {
    const int m   = blockIdx.x;               // 0..2047
    const int b   = m >> 10;                  // / (N*L = 1024)
    const int rem = m & 1023;
    const int n   = rem >> 2;
    const int l   = rem & 3;
    long long ctx = anchor[b * N_ + n] - 1;
    if (ctx < 0) ctx = 0;
    const float* src = hs + (((size_t)l * B_ + b) * S_ + (size_t)ctx) * H_;
    unsigned short* dst = A + (size_t)m * GK;
    for (int i = threadIdx.x; i < GK; i += blockDim.x)
        dst[i] = f32_to_bf16_rne(src[i]);
}

// ---------------------------------------------------------------------------
// K3: bf16 WMMA GEMM: C[GM][GN] = A[GM][GK] * Wt^T
//     block = 256 threads (8 waves), tile 128x128, wave tile 64x32,
//     K-step 32 = one v_wmma_f32_16x16x32_bf16 per 16x16 tile.
//     Double-buffered LDS fed by GLOBAL_LOAD_ASYNC_TO_LDS_B128 (ASYNCcnt):
//     4 async instrs/wave/tile; s_wait_asynccnt 4 waits only for the tile
//     about to be consumed, so next-tile HBM fetches overlap the WMMAs.
// ---------------------------------------------------------------------------
__global__ __launch_bounds__(256)
void gemm_bf16_wmma_kernel(const unsigned short* __restrict__ A_,
                           const unsigned short* __restrict__ Wt_,
                           float* __restrict__ C) {
    const __bf16* __restrict__ A  = reinterpret_cast<const __bf16*>(A_);
    const __bf16* __restrict__ Wt = reinterpret_cast<const __bf16*>(Wt_);

    __shared__ __bf16 Asm[2][128 * 32];       // [buf][row][k], k contiguous
    __shared__ __bf16 Bsm[2][128 * 32];       // [buf][col][k], k contiguous

    const int mBase = blockIdx.y * 128;
    const int cBase = blockIdx.x * 128;
    const int tid   = threadIdx.x;
    const int wave  = tid >> 5;
    const int lane  = tid & 31;
    const int wm    = wave >> 2;              // 0..1 : 64-row strip
    const int wn    = wave & 3;               // 0..3 : 32-col strip
    const int lr    = lane & 15;
    const int lh    = lane >> 4;

    // Issue the 4 async global->LDS B128 copies this thread owns for one tile.
    auto stage = [&](int p, int kk) {
#pragma unroll
        for (int t = 0; t < 2; ++t) {
            int c   = tid + t * 256;
            int row = c >> 2;
            int cc  = c & 3;
            unsigned lA = (unsigned)(size_t)&Asm[p][row * 32 + cc * 8];
            unsigned lB = (unsigned)(size_t)&Bsm[p][row * 32 + cc * 8];
            const __bf16* gA = A  + (size_t)(mBase + row) * GK + kk + cc * 8;
            const __bf16* gB = Wt + (size_t)(cBase + row) * GK + kk + cc * 8;
            asm volatile("global_load_async_to_lds_b128 %0, %1, off"
                         :: "v"(lA), "v"(gA) : "memory");
            asm volatile("global_load_async_to_lds_b128 %0, %1, off"
                         :: "v"(lB), "v"(gB) : "memory");
        }
    };

    v8f acc[4][2] = {};

    stage(0, 0);                              // prologue: fill buffer 0

    for (int k0 = 0; k0 < GK; k0 += 32) {
        const int p = (k0 >> 5) & 1;
        if (k0 + 32 < GK) {
            stage(1 - p, k0 + 32);            // prefetch next tile (async)
            asm volatile("s_wait_asynccnt 0x4" ::: "memory");  // current tile landed
        } else {
            asm volatile("s_wait_asynccnt 0x0" ::: "memory");
        }
        __syncthreads();                      // all waves' portions visible

        // ---- A fragments: 16x32 bf16, contiguous K run (matches B):
        // lanes 0-15: K0..15 ; lanes 16-31: K16..31
        v16bf afrag[4];
#pragma unroll
        for (int i = 0; i < 4; ++i) {
            const __bf16* pa = &Asm[p][(wm * 64 + i * 16 + lr) * 32];
            v8bf lo = *reinterpret_cast<const v8bf*>(pa + 16 * lh);
            v8bf hi = *reinterpret_cast<const v8bf*>(pa + 16 * lh + 8);
            afrag[i] = __builtin_shufflevector(lo, hi,
                        0,1,2,3,4,5,6,7,8,9,10,11,12,13,14,15);
        }
        // ---- B fragments: 32x16 bf16, lane = column, contiguous K run
        v16bf bfrag[2];
#pragma unroll
        for (int j = 0; j < 2; ++j) {
            const __bf16* pb = &Bsm[p][(wn * 32 + j * 16 + lr) * 32 + 16 * lh];
            bfrag[j] = *reinterpret_cast<const v16bf*>(pb);
        }
        // ---- 8 WMMAs per wave per K-step ----
#pragma unroll
        for (int i = 0; i < 4; ++i)
#pragma unroll
            for (int j = 0; j < 2; ++j)
                acc[i][j] = __builtin_amdgcn_wmma_f32_16x16x32_bf16(
                    false, afrag[i], false, bfrag[j],
                    (short)0, acc[i][j], false, false);
        __syncthreads();                      // buffer p free for reuse
    }

    // ---- store: C/D layout: row = e + 8*lh, col = lr ----
#pragma unroll
    for (int i = 0; i < 4; ++i)
#pragma unroll
        for (int j = 0; j < 2; ++j)
#pragma unroll
            for (int e = 0; e < 8; ++e) {
                int row = mBase + wm * 64 + i * 16 + e + 8 * lh;
                int col = cBase + wn * 32 + j * 16 + lr;
                C[(size_t)row * GN + col] = acc[i][j][e];
            }
}

// ---------------------------------------------------------------------------
// K4: attention per (b,n): scores (16x20 per head), softmax, PV, mask, store.
// ---------------------------------------------------------------------------
__global__ __launch_bounds__(512)
void attn_kernel(const float* __restrict__ q,
                 const float* __restrict__ k,
                 const float* __restrict__ v,
                 const float* __restrict__ kvC,     // GEMM output [GM][GN]
                 const unsigned char* __restrict__ keep,
                 float* __restrict__ out) {
    __shared__ float sc[NH_][BS_][20];
    const int b = blockIdx.x >> 8;
    const int n = blockIdx.x & 255;
    const int tid = threadIdx.x;
    const size_t tokBase = (size_t)b * (N_ * BS_) + (size_t)n * BS_;
    const size_t rowBase = (size_t)(b * N_ + n) * L_;

    // Phase 1: scores = q . k_ext * SCALE   (5120 dots / 512 threads)
    for (int t = tid; t < NH_ * BS_ * 20; t += 512) {
        int j  = t % 20;
        int rr = (t / 20) % BS_;
        int h  = t / (20 * BS_);
        const float* qp = q + ((tokBase + rr) * NH_ + h) * DH_;
        const float* kp = (j < L_)
            ? (kvC + (rowBase + j) * GN + h * DH_)                    // k_chs
            : (k + ((tokBase + (j - L_)) * NH_ + h) * DH_);           // kb
        const float4* q4 = reinterpret_cast<const float4*>(qp);
        const float4* k4 = reinterpret_cast<const float4*>(kp);
        float acc = 0.f;
#pragma unroll 8
        for (int d = 0; d < DH_ / 4; ++d) {
            float4 a = q4[d], bb = k4[d];
            acc += a.x * bb.x + a.y * bb.y + a.z * bb.z + a.w * bb.w;
        }
        sc[h][rr][j] = acc * SCALE_;
    }
    __syncthreads();

    // Phase 2: softmax over 20 keys (256 rows)
    if (tid < NH_ * BS_) {
        int h = tid >> 4, rr = tid & 15;
        float m = sc[h][rr][0];
#pragma unroll
        for (int j = 1; j < 20; ++j) m = fmaxf(m, sc[h][rr][j]);
        float e[20], s = 0.f;
#pragma unroll
        for (int j = 0; j < 20; ++j) { e[j] = __expf(sc[h][rr][j] - m); s += e[j]; }
        float inv = 1.f / s;
#pragma unroll
        for (int j = 0; j < 20; ++j) sc[h][rr][j] = e[j] * inv;
    }
    __syncthreads();

    // Phase 3: out = attn @ v_ext, gated by block_keep_mask
    const float gate = keep[b * N_ + n] ? 1.f : 0.f;
    for (int t = tid; t < NH_ * BS_ * DH_; t += 512) {
        int d  = t & (DH_ - 1);
        int rr = (t >> 7) & 15;
        int h  = t >> 11;
        float acc = 0.f;
#pragma unroll
        for (int j = 0; j < 20; ++j) {
            const float* vp = (j < L_)
                ? (kvC + (rowBase + j) * GN + 2048 + h * DH_ + d)     // v_chs
                : (v + ((tokBase + (j - L_)) * NH_ + h) * DH_ + d);   // vb
            acc += sc[h][rr][j] * (*vp);
        }
        out[(tokBase + rr) * (NH_ * DH_) + h * DH_ + d] = acc * gate;
    }
}

// ---------------------------------------------------------------------------
extern "C" void kernel_launch(void* const* d_in, const int* in_sizes, int n_in,
                              void* d_out, int out_size, void* d_ws, size_t ws_size,
                              hipStream_t stream) {
    const float*         hs     = (const float*)d_in[0];
    const long long*     anchor = (const long long*)d_in[1];
    const unsigned char* keep   = (const unsigned char*)d_in[2];
    const float*         q      = (const float*)d_in[3];
    const float*         k      = (const float*)d_in[4];
    const float*         v      = (const float*)d_in[5];
    const float*         Wkc    = (const float*)d_in[6];
    const float*         Wvc    = (const float*)d_in[7];

    char* ws = (char*)d_ws;
    unsigned short* Abf = (unsigned short*)(ws);                        //  8 MB: A bf16 [2048][2048]
    unsigned short* Wt  = (unsigned short*)(ws + ((size_t)8  << 20));   // 16 MB: Wt bf16 [4096][2048]
    float*          Cws = (float*)        (ws + ((size_t)24 << 20));    // 32 MB: C f32 [2048][4096]

    wt_transpose_kernel<<<dim3(GN / 32, GK / 32), dim3(32, 8), 0, stream>>>(Wkc, Wvc, Wt);
    gather_a_kernel<<<GM, 256, 0, stream>>>(hs, anchor, Abf);
    gemm_bf16_wmma_kernel<<<dim3(GN / 128, GM / 128), 256, 0, stream>>>(Abf, Wt, Cws);
    attn_kernel<<<B_ * N_, 512, 0, stream>>>(q, k, v, Cws, keep, (float*)d_out);
}
